// SparseMoELayer_66546223284321
// MI455X (gfx1250) — matrix operand order repaired and checked by reading, not compiled, not run on hardware
//
#include <hip/hip_runtime.h>
#include <hip/hip_bf16.h>

#define H        1024
#define DFF      4096
#define E_ROUTED 7
#define T_TOK    8192
#define TILE_M   32    // tokens per block (2 m-tiles)
#define CHUNK    128   // dff columns per block iteration
#define THREADS  512   // 16 waves

typedef __attribute__((ext_vector_type(16))) __bf16        v16bf;
typedef __attribute__((ext_vector_type(2)))  __bf16        v2bf;
typedef __attribute__((ext_vector_type(8)))  float         v8f;
typedef __attribute__((ext_vector_type(4)))  unsigned int  v4u;
typedef __attribute__((ext_vector_type(8)))  int           v8i;
typedef __attribute__((ext_vector_type(4)))  int           v4i;

#if __has_builtin(__builtin_amdgcn_tensor_load_to_lds) && \
    __has_builtin(__builtin_amdgcn_s_wait_tensorcnt)
#define USE_TDM 1
#else
#define USE_TDM 0
#endif

union FragU {
    unsigned int u[8];
    v16bf b;
};

#if __has_builtin(__builtin_amdgcn_cvt_pk_bf16_f32)
__device__ __forceinline__ unsigned int pack_bf16(float lo, float hi) {
    union { v2bf b; unsigned int u; } c;
    c.b = __builtin_amdgcn_cvt_pk_bf16_f32(lo, hi);
    return c.u;
}
#else
__device__ __forceinline__ unsigned int bf16_rne(float f) {
    unsigned int a = __float_as_uint(f);
    return (a + 0x7FFFu + ((a >> 16) & 1u)) >> 16;
}
__device__ __forceinline__ unsigned int pack_bf16(float lo, float hi) {
    return (bf16_rne(lo) & 0xFFFFu) | (bf16_rne(hi) << 16);
}
#endif

// ---------------------------------------------------------------- utilities
__global__ void moe_zero_kernel(int* counts) {
    if (threadIdx.x < E_ROUTED) counts[threadIdx.x] = 0;
}

// f32 pairs -> packed bf16 dwords (x and, ws permitting, all weights)
__global__ void moe_cvt_kernel(const float* __restrict__ x,
                               unsigned int* __restrict__ xb, int nPairs) {
    int i = blockIdx.x * blockDim.x + threadIdx.x;
    if (i < nPairs) {
        float2 v = ((const float2*)x)[i];
        xb[i] = pack_bf16(v.x, v.y);
    }
}

// ---------------------------------------------------------------- router
__global__ __launch_bounds__(256)
void moe_router_kernel(const float* __restrict__ x, const float* __restrict__ rw,
                       int* __restrict__ counts, int* __restrict__ idxL,
                       float* __restrict__ wgtL) {
    const int lane = threadIdx.x & 31;
    const int wv   = threadIdx.x >> 5;
    const int t    = blockIdx.x * 8 + wv;

    float p[E_ROUTED];
#pragma unroll
    for (int e = 0; e < E_ROUTED; ++e) p[e] = 0.f;

    for (int h = lane; h < H; h += 32) {
        float xv = x[(size_t)t * H + h];
#pragma unroll
        for (int e = 0; e < E_ROUTED; ++e) p[e] += xv * rw[e * H + h];
    }
#pragma unroll
    for (int e = 0; e < E_ROUTED; ++e) {
#pragma unroll
        for (int off = 16; off > 0; off >>= 1)
            p[e] += __shfl_xor(p[e], off, 32);
    }
    if (lane == 0) {
        float mx = p[0];
#pragma unroll
        for (int e = 1; e < E_ROUTED; ++e) mx = fmaxf(mx, p[e]);
        float s = 0.f;
#pragma unroll
        for (int e = 0; e < E_ROUTED; ++e) { p[e] = __expf(p[e] - mx); s += p[e]; }
        float inv = __builtin_amdgcn_rcpf(s);
#pragma unroll
        for (int e = 0; e < E_ROUTED; ++e) p[e] *= inv;

        int e1 = 0;
#pragma unroll
        for (int e = 1; e < E_ROUTED; ++e) if (p[e] > p[e1]) e1 = e;
        int e2 = (e1 == 0) ? 1 : 0;
#pragma unroll
        for (int e = 0; e < E_ROUTED; ++e) if (e != e1 && p[e] > p[e2]) e2 = e;

        float v1 = p[e1], v2 = p[e2];
        float inv2 = __builtin_amdgcn_rcpf(v1 + v2 + 1e-9f);
        int s1 = atomicAdd(&counts[e1], 1);
        idxL[e1 * T_TOK + s1] = t;
        wgtL[e1 * T_TOK + s1] = v1 * inv2;
        int s2 = atomicAdd(&counts[e2], 1);
        idxL[e2 * T_TOK + s2] = t;
        wgtL[e2 * T_TOK + s2] = v2 * inv2;
    }
}

// -------------------------------------------------- fragment loaders (wave32)
// A-matrix 16x32 bf16 (this code's contiguous k order, as in B): lane m=L&15, kh=L>>4;
//   u[g*4+p] holds K = kbase + kh*16 + g*8 + p*2 (pair) -> 2x ds_load_b128
__device__ __forceinline__ v16bf load_fragA_lds(const unsigned int* lds,
                                                int kbaseHalf, int strideU) {
    const int lane = threadIdx.x & 31;
    const int m = lane & 15, kh = lane >> 4;
    FragU f;
#pragma unroll
    for (int g = 0; g < 2; ++g)
#pragma unroll
        for (int p = 0; p < 4; ++p)
            f.u[g * 4 + p] = lds[m * strideU + kbaseHalf + kh * 8 + g * 4 + p];
    return f.b;
}

// B-matrix 32x16 bf16: lane n=L&15 (column = W row), kh=L>>4;
//   u[v] holds K = kbase + kh*16 + v*2.  bf16 weights: 2x global_load_b128.
__device__ __forceinline__ v16bf load_fragB_bf16(const unsigned int* __restrict__ rowU,
                                                 int kbaseHalf) {
    const int kh = (threadIdx.x >> 4) & 1;
    const unsigned int* p = rowU + kbaseHalf + kh * 8;
    FragU f;
#pragma unroll
    for (int v = 0; v < 8; ++v) f.u[v] = p[v];
    return f.b;
}

// f32-weight fallback: convert in registers (only used if ws too small)
__device__ __forceinline__ v16bf load_fragB_f32(const float* __restrict__ wRow,
                                                int kbase) {
    const int kh = (threadIdx.x >> 4) & 1;
    FragU f;
#pragma unroll
    for (int v = 0; v < 8; ++v) {
        float2 w2 = *(const float2*)(wRow + kbase + kh * 16 + v * 2);
        f.u[v] = pack_bf16(w2.x, w2.y);
    }
    return f.b;
}

#if USE_TDM
// TDM: 1-D tile load, nElem 4-byte elements, global -> LDS (D# per ISA 8.3/8.4)
__device__ __forceinline__ void tdm_load_row(unsigned int ldsByteAddr,
                                             unsigned long long gaddr,
                                             unsigned int nElem) {
    v4u g0;
    g0[0] = 1u;                                         // count=1, user mode
    g0[1] = ldsByteAddr;                                // lds_addr
    g0[2] = (unsigned int)gaddr;                        // global_addr[31:0]
    g0[3] = (unsigned int)(gaddr >> 32) | (2u << 30);   // addr[56:32] | type=2
    v8i g1;
    g1[0] = (int)(2u << 16);                            // data_size = 4 bytes
    g1[1] = (int)((nElem & 0xFFFFu) << 16);             // tensor_dim0[15:0]
    g1[2] = (int)((nElem >> 16) | (1u << 16));          // tensor_dim0 hi | tensor_dim1=1
    g1[3] = (int)((nElem & 0xFFFFu) << 16);             // tile_dim0
    g1[4] = 0; g1[5] = (int)nElem;                      // tensor_dim0_stride
    g1[6] = 0; g1[7] = 0;
    v4i z4 = {0, 0, 0, 0};
#if __clang_major__ >= 23
    v8i z8 = {0, 0, 0, 0, 0, 0, 0, 0};
    __builtin_amdgcn_tensor_load_to_lds(g0, g1, z4, z4, z8, 0);
#else
    __builtin_amdgcn_tensor_load_to_lds(g0, g1, z4, z4, 0);
#endif
}
#endif

// ---------------------------------------------------------------- expert GEMM
// Block: 32 tokens x 1 expert, 16 waves.
//   wave -> m-tile mh = wv&1 (rows mh*16..+16), column slice wq = wv>>1.
// MODE  0: shared expert (identity tokens, weight 1, plain store -> inits out)
// MODE  1: routed expert e=blockIdx.y (gathered tokens, atomicAdd)
// WMODE 0: f32 weights, convert on the fly   WMODE 1: pre-converted bf16 weights
template <int MODE, int WMODE>
__global__ __launch_bounds__(THREADS)
void moe_expert_kernel(const unsigned int* __restrict__ xb,   // [T, H/2] packed bf16
                       const float* __restrict__ WgF, const float* __restrict__ WuF,
                       const float* __restrict__ WdF,
                       const unsigned int* __restrict__ WgB,
                       const unsigned int* __restrict__ WuB,
                       const unsigned int* __restrict__ WdB,
                       const int* __restrict__ counts,
                       const int* __restrict__ idxL,
                       const float* __restrict__ wgtL,
                       float* __restrict__ out) {
    __shared__ unsigned int sA[TILE_M * (H / 2)];      // 64 KB token tile (bf16)
    __shared__ unsigned int sH[TILE_M * (CHUNK / 2)];  // 8 KB  h tile (bf16)
    __shared__ int   sTok[TILE_M];
    __shared__ float sWgt[TILE_M];

    const int e    = blockIdx.y;
    const int tile = blockIdx.x;
    const int tid  = threadIdx.x;
    const int lane = tid & 31;
    const int wv   = tid >> 5;
    const int mh   = wv & 1;        // which 16-row m-tile
    const int wq   = wv >> 1;       // column-slice index 0..7
    const int n    = lane & 15;
    const int kh   = lane >> 4;

    int cnt = 0;
    if constexpr (MODE == 1) {
        cnt = counts[e];
        if (tile * TILE_M >= cnt) return;
    }

    const float *Wg = WgF, *Wu = WuF, *Wd = WdF;
    const unsigned int *WgU = WgB, *WuU = WuB, *WdU = WdB;
    if constexpr (MODE == 1) {
        if constexpr (WMODE == 0) {
            Wg = WgF + (size_t)e * DFF * H;
            Wu = WuF + (size_t)e * DFF * H;
            Wd = WdF + (size_t)e * H * DFF;
        } else {
            WgU = WgB + (size_t)e * DFF * (H / 2);
            WuU = WuB + (size_t)e * DFF * (H / 2);
            WdU = WdB + (size_t)e * H * (DFF / 2);
        }
    }

    if (tid < TILE_M) {
        int g = tile * TILE_M + tid;
        if constexpr (MODE == 0) { sTok[tid] = g; sWgt[tid] = 1.0f; }
        else {
            int gi = (g < cnt) ? g : (cnt - 1);
            sTok[tid] = idxL[e * T_TOK + gi];
            sWgt[tid] = (g < cnt) ? wgtL[e * T_TOK + g] : 0.0f;
        }
    }
    __syncthreads();

    // ---- gather 32 token rows (bf16) into LDS; reused across all 32 chunks
#if USE_TDM
    if (wv == 0) {
        const unsigned int sABase = (unsigned int)(unsigned long long)(uintptr_t)&sA[0];
#pragma unroll
        for (int m = 0; m < TILE_M; ++m) {
            unsigned long long g =
                (unsigned long long)(uintptr_t)(xb + (size_t)sTok[m] * (H / 2));
            tdm_load_row(sABase + m * (H / 2) * 4u, g, H / 2);
        }
        __builtin_amdgcn_s_wait_tensorcnt(0);
    }
    __syncthreads();
#else
    for (int i = tid; i < TILE_M * (H / 2); i += THREADS) {
        int m = i >> 9;         // H/2 == 512
        int c = i & 511;
        sA[i] = xb[(size_t)sTok[m] * (H / 2) + c];
    }
    __syncthreads();
#endif

    const unsigned int* sAm = sA + mh * 16 * (H / 2);       // this wave's m-tile
    const unsigned int* sHm = sH + mh * 16 * (CHUNK / 2);

    v8f acc[8];                 // out rows mh*16.., columns [wq*128, wq*128+128)
#pragma unroll
    for (int j = 0; j < 8; ++j) acc[j] = v8f{};

    for (int c = 0; c < DFF / CHUNK; ++c) {
        const int chunkBase = c * CHUNK;
        const int dcol = chunkBase + wq * 16 + n;          // this lane's dff column

        const float *wgRow = nullptr, *wuRow = nullptr;
        const unsigned int *wgRowU = nullptr, *wuRowU = nullptr;
        if constexpr (WMODE == 0) {
            wgRow = Wg + (size_t)dcol * H;
            wuRow = Wu + (size_t)dcol * H;
            if (c + 1 < DFF / CHUNK) {                     // global_prefetch_b8
                __builtin_prefetch(wgRow + (size_t)CHUNK * H, 0, 1);
                __builtin_prefetch(wuRow + (size_t)CHUNK * H, 0, 1);
            }
        } else {
            wgRowU = WgU + (size_t)dcol * (H / 2);
            wuRowU = WuU + (size_t)dcol * (H / 2);
            if (c + 1 < DFF / CHUNK) {
                __builtin_prefetch(wgRowU + (size_t)CHUNK * (H / 2), 0, 1);
                __builtin_prefetch(wuRowU + (size_t)CHUNK * (H / 2), 0, 1);
            }
        }

        // ---- phase A: gate & up, M16(m-tile) x N16(per wave) x K1024
        v8f cg{}, cu{};
#pragma unroll 4
        for (int ks = 0; ks < H / 32; ++ks) {
            const int kbase = ks * 32;
            v16bf fa = load_fragA_lds(sAm, kbase >> 1, H / 2);
            v16bf fg, fu;
            if constexpr (WMODE == 0) {
                fg = load_fragB_f32(wgRow, kbase);
                fu = load_fragB_f32(wuRow, kbase);
            } else {
                fg = load_fragB_bf16(wgRowU, kbase >> 1);
                fu = load_fragB_bf16(wuRowU, kbase >> 1);
            }
            cg = __builtin_amdgcn_wmma_f32_16x16x32_bf16(false, fa, false, fg,
                                                         (short)0, cg, false, false);
            cu = __builtin_amdgcn_wmma_f32_16x16x32_bf16(false, fa, false, fu,
                                                         (short)0, cu, false, false);
        }

        __syncthreads();   // previous chunk's phase B finished reading sH
        {
            unsigned short* sHh = (unsigned short*)sH;
#pragma unroll
            for (int r = 0; r < 8; ++r) {
                int m = mh * 16 + kh * 8 + r;              // C/D row for VGPR r
                float g = cg[r], u = cu[r];
                float hv = g * __builtin_amdgcn_rcpf(1.0f + __expf(-g)) * u;
                sHh[m * CHUNK + wq * 16 + n] = (unsigned short)pack_bf16(hv, 0.f);
            }
        }
        __syncthreads();

        // ---- phase B: down, M16(m-tile) x N128(per wave) x K128
#pragma unroll
        for (int kk = 0; kk < 4; ++kk) {
            v16bf fh = load_fragA_lds(sHm, kk * 16, CHUNK / 2);
#pragma unroll
            for (int j = 0; j < 8; ++j) {
                const int colW = wq * 128 + j * 16 + n;    // out column (row of Wd)
                v16bf fd;
                if constexpr (WMODE == 0) {
                    fd = load_fragB_f32(Wd + (size_t)colW * DFF + chunkBase, kk * 32);
                } else {
                    fd = load_fragB_bf16(WdU + (size_t)colW * (DFF / 2) + chunkBase / 2,
                                         kk * 16);
                }
                acc[j] = __builtin_amdgcn_wmma_f32_16x16x32_bf16(false, fh, false, fd,
                                                                 (short)0, acc[j],
                                                                 false, false);
            }
        }
    }

    // ---- epilogue: scale by combine weight, write/accumulate
#pragma unroll
    for (int j = 0; j < 8; ++j) {
        const int col = wq * 128 + j * 16 + n;
#pragma unroll
        for (int r = 0; r < 8; ++r) {
            int m = mh * 16 + kh * 8 + r;
            float val = acc[j][r] * sWgt[m];
            float* dst = out + (size_t)sTok[m] * H + col;
            if constexpr (MODE == 0) *dst = val;
            else                     atomicAdd(dst, val);
        }
    }
}

// ---------------------------------------------------------------- launcher
extern "C" void kernel_launch(void* const* d_in, const int* in_sizes, int n_in,
                              void* d_out, int out_size, void* d_ws, size_t ws_size,
                              hipStream_t stream) {
    const float* x           = (const float*)d_in[0];
    const float* shared_up   = (const float*)d_in[1];
    const float* shared_gate = (const float*)d_in[2];
    const float* shared_down = (const float*)d_in[3];
    const float* routed_up   = (const float*)d_in[4];
    const float* routed_gate = (const float*)d_in[5];
    const float* routed_down = (const float*)d_in[6];
    const float* router_w    = (const float*)d_in[7];
    float* out = (float*)d_out;

    char* ws = (char*)d_ws;
    int*          counts = (int*)ws;                                   // [7]
    int*          idxL   = (int*)(ws + 256);                           // [7*8192]
    float*        wgtL   = (float*)(ws + 256 + E_ROUTED * T_TOK * 4);  // [7*8192]
    unsigned int* xb     = (unsigned int*)(ws + 459264);               // 16 MB

    const size_t SW  = (size_t)DFF * H / 2;       // uints per shared weight matrix
    const size_t RW  = (size_t)E_ROUTED * SW;     // uints per routed weight tensor
    const size_t WB_OFF = 459264 + (size_t)T_TOK * (H / 2) * 4;        // 17236480
    unsigned int* wb   = (unsigned int*)(ws + WB_OFF);
    unsigned int* w_sg = wb;
    unsigned int* w_su = wb + SW;
    unsigned int* w_sd = wb + 2 * SW;
    unsigned int* w_rg = wb + 3 * SW;
    unsigned int* w_ru = w_rg + RW;
    unsigned int* w_rd = w_ru + RW;
    const size_t needW = WB_OFF + (3 * SW + 3 * RW) * 4;               // ~208 MiB
    const bool bf16W = (ws_size >= needW);

    moe_zero_kernel<<<1, 32, 0, stream>>>(counts);

    const int nPx = T_TOK * (H / 2);
    moe_cvt_kernel<<<(nPx + 255) / 256, 256, 0, stream>>>(x, xb, nPx);

    moe_router_kernel<<<T_TOK / 8, 256, 0, stream>>>(x, router_w, counts, idxL, wgtL);

    if (bf16W) {
        const int nPs = (int)SW, nPr = (int)RW;
        moe_cvt_kernel<<<(nPs + 255) / 256, 256, 0, stream>>>(shared_gate, w_sg, nPs);
        moe_cvt_kernel<<<(nPs + 255) / 256, 256, 0, stream>>>(shared_up,   w_su, nPs);
        moe_cvt_kernel<<<(nPs + 255) / 256, 256, 0, stream>>>(shared_down, w_sd, nPs);
        moe_cvt_kernel<<<(nPr + 255) / 256, 256, 0, stream>>>(routed_gate, w_rg, nPr);
        moe_cvt_kernel<<<(nPr + 255) / 256, 256, 0, stream>>>(routed_up,   w_ru, nPr);
        moe_cvt_kernel<<<(nPr + 255) / 256, 256, 0, stream>>>(routed_down, w_rd, nPr);

        moe_expert_kernel<0, 1><<<dim3(T_TOK / TILE_M, 1), THREADS, 0, stream>>>(
            xb, nullptr, nullptr, nullptr, w_sg, w_su, w_sd, counts, idxL, wgtL, out);
        moe_expert_kernel<1, 1><<<dim3(T_TOK / TILE_M, E_ROUTED), THREADS, 0, stream>>>(
            xb, nullptr, nullptr, nullptr, w_rg, w_ru, w_rd, counts, idxL, wgtL, out);
    } else {
        moe_expert_kernel<0, 0><<<dim3(T_TOK / TILE_M, 1), THREADS, 0, stream>>>(
            xb, shared_gate, shared_up, shared_down, nullptr, nullptr, nullptr,
            counts, idxL, wgtL, out);
        moe_expert_kernel<1, 0><<<dim3(T_TOK / TILE_M, E_ROUTED), THREADS, 0, stream>>>(
            xb, routed_gate, routed_up, routed_down, nullptr, nullptr, nullptr,
            counts, idxL, wgtL, out);
    }
}
